// PifPredictor_85237920956497
// MI455X (gfx1250) — matrix operand order repaired and checked by reading, not compiled, not run on hardware
//
#include <hip/hip_runtime.h>
#include <hip/hip_bf16.h>
#include <math.h>

// ---- problem constants (match reference) ----
constexpr int NB = 16;      // batch
constexpr int NT = 2048;    // time
constexpr int NC = 512;     // channels
constexpr int NH = 4;       // heads
constexpr int NU = 250;     // max_token
constexpr int ND = NC / NH; // 128 head dim
constexpr float SMOOTH = 1.0f;
constexpr float NOISE  = 0.0f;

typedef __attribute__((ext_vector_type(16))) __bf16 v16bf;
typedef __attribute__((ext_vector_type(2)))  __bf16 bf2;
typedef __attribute__((ext_vector_type(8)))  float  v8f;

// -------------------------------------------------------------------------
// Kernel 1: alpha prediction branch. One wave32 per (b,t).
// -------------------------------------------------------------------------
__global__ __launch_bounds__(256) void alpha_kernel(
    const float* __restrict__ hidden, const float* __restrict__ mask,
    const float* __restrict__ conv_w, const float* __restrict__ conv_b,
    const float* __restrict__ lin_w,  const float* __restrict__ lin_b,
    float* __restrict__ raw)
{
    int gw   = (blockIdx.x * 256 + threadIdx.x) >> 5;
    int lane = threadIdx.x & 31;
    if (gw >= NB * NT) return;
    int b = gw / NT, t = gw % NT;

    float partial = 0.f;
    for (int c = lane; c < NC; c += 32) {
        float center = hidden[(b * NT + t) * NC + c];
        float acc = conv_b[c] + center;            // conv + residual
#pragma unroll
        for (int k = 0; k < 3; ++k) {
            int tt = t + k - 1;                    // L=1 left pad
            float hv = (tt >= 0 && tt < NT) ? hidden[(b * NT + tt) * NC + c] : 0.f;
            acc = fmaf(conv_w[c * 3 + k], hv, acc);
        }
        acc = fmaxf(acc, 0.f);                     // relu
        partial = fmaf(acc, lin_w[c], partial);    // Linear(C->1)
    }
#pragma unroll
    for (int off = 16; off > 0; off >>= 1)
        partial += __shfl_xor(partial, off, 32);

    if (lane == 0) {
        float x = partial + lin_b[0];
        float a = 1.f / (1.f + expf(-x));
        a = fmaxf(a * SMOOTH - NOISE, 0.f);
        raw[b * NT + t] = a * mask[b * NT + t];
    }
}

// -------------------------------------------------------------------------
// Kernel 2: per-batch reduce + scale + inclusive scan. One block per batch.
// -------------------------------------------------------------------------
__global__ __launch_bounds__(256) void scan_kernel(
    const float* __restrict__ raw, const int* __restrict__ tgt_len,
    float* __restrict__ align, float* __restrict__ out_token,
    float* __restrict__ out_alphas)
{
    int b = blockIdx.x, tid = threadIdx.x;
    __shared__ float ssum[256];

    float v[8]; float s = 0.f;
#pragma unroll
    for (int j = 0; j < 8; ++j) { v[j] = raw[b * NT + tid * 8 + j]; s += v[j]; }
    ssum[tid] = s;
    __syncthreads();

    for (int off = 1; off < 256; off <<= 1) {
        float add = (tid >= off) ? ssum[tid - off] : 0.f;
        __syncthreads();
        ssum[tid] += add;
        __syncthreads();
    }
    float total = ssum[255];
    float excl  = ssum[tid] - s;
    float scale = (float)tgt_len[b] / total;

    float run = excl;
#pragma unroll
    for (int j = 0; j < 8; ++j) {
        run += v[j];
        align[b * NT + tid * 8 + j]      = run * scale;
        out_alphas[b * NT + tid * 8 + j] = v[j] * scale;
    }
    if (tid == 0) out_token[b] = total;
}

// -------------------------------------------------------------------------
// Kernel 3: per-row softmax stats (max, 1/sumexp). One wave per (b,h,u).
// -------------------------------------------------------------------------
__global__ __launch_bounds__(256) void rowstat_kernel(
    const float* __restrict__ align, const float* __restrict__ mask,
    const float* __restrict__ sigma, const float* __restrict__ sigma_bias,
    float* __restrict__ mrow, float* __restrict__ isrow)
{
    int gw = (blockIdx.x * 256 + threadIdx.x) >> 5;
    if (gw >= NB * NH * NU) return;
    int lane = threadIdx.x & 31;
    int b = gw / (NH * NU); int rem = gw % (NH * NU);
    int h = rem / NU;       int u   = rem % NU;
    float sig = sigma[h], sb = sigma_bias[h];
    float uf = (float)u + 0.5f;

    float m = -INFINITY;
    for (int t = lane; t < NT; t += 32) {
        float d  = (uf - align[b * NT + t]) * sig;
        float sc = -(d * d) + sb;
        if (mask[b * NT + t] > 0.f) m = fmaxf(m, sc);
    }
#pragma unroll
    for (int off = 16; off > 0; off >>= 1)
        m = fmaxf(m, __shfl_xor(m, off, 32));
    if (m == -INFINITY) m = 0.f;

    float s = 0.f;
    for (int t = lane; t < NT; t += 32) {
        float d  = (uf - align[b * NT + t]) * sig;
        float sc = -(d * d) + sb;
        if (mask[b * NT + t] > 0.f) s += expf(sc - m);
    }
#pragma unroll
    for (int off = 16; off > 0; off >>= 1)
        s += __shfl_xor(s, off, 32);

    if (lane == 0) { mrow[gw] = m; isrow[gw] = (s > 0.f) ? 1.f / s : 0.f; }
}

// -------------------------------------------------------------------------
// Kernel 4: WMMA attention einsum. Block = (b, h, u-tile of 16), 8 waves.
// K-chunk = 64: two independent v_wmma_f32_16x16x32_bf16 per wave per chunk.
// LDS tiles are staged directly in WMMA fragment layout so each lane reads
// its v16bf fragment as one contiguous 32B access (ds_load_b128 pair).
//   A (16-bit 16x32): lane = M + 16*((K>>3)&1), elem = (K&7) + 8*(K>>4)
//   B (16-bit 32x16): lane = N + 16*((K>>3)&1), elem = (K&7) + 8*(K>>4)
// -------------------------------------------------------------------------
__global__ __launch_bounds__(256) void attn_wmma_kernel(
    const float* __restrict__ hidden, const float* __restrict__ mask,
    const float* __restrict__ align,
    const float* __restrict__ mrow,  const float* __restrict__ isrow,
    const float* __restrict__ sigma, const float* __restrict__ sigma_bias,
    float* __restrict__ out /* acoustic [B,U,C] */)
{
    int blk = blockIdx.x;                          // b*NH*16 + h*16 + ut
    int ut = blk & 15, h = (blk >> 4) & (NH - 1), b = blk >> 6;
    int u0 = ut * 16;
    int tid = threadIdx.x, wid = tid >> 5, lane = tid & 31;

    __shared__ __align__(32) __bf16 frag_b[8][2][32][16]; // [wave][kchunk][lane][e] 16KB
    __shared__ __align__(32) __bf16 frag_a[2][32][16];    // [kchunk][lane][e]       2KB
    __shared__ __align__(16) float  align_s[64];
    __shared__ __align__(16) float  mask_s[64];
    __shared__ float m_s[16], is_s[16];

    if (tid < 16) {
        int u = u0 + tid;
        float m = 0.f, is = 0.f;
        if (u < NU) { int r = (b * NH + h) * NU + u; m = mrow[r]; is = isrow[r]; }
        m_s[tid] = m; is_s[tid] = is;              // is=0 for pad rows -> weight 0
    }
    float sig = sigma[h], sb = sigma_bias[h];
    v8f acc0 = {}, acc1 = {};

    for (int t0 = 0; t0 < NT; t0 += 64) {
        // ---- phase 1: async-stage align/mask chunk; stage V fragments ----
        if (tid < 32) {
            const float* gsrc = (tid < 16) ? &align[b * NT + t0 + (tid & 15) * 4]
                                           : &mask [b * NT + t0 + (tid & 15) * 4];
            float* ldst_p = (tid < 16) ? &align_s[(tid & 15) * 4]
                                       : &mask_s [(tid & 15) * 4];
            unsigned int ldst = (unsigned int)(unsigned long long)ldst_p; // LDS offset
            asm volatile("global_load_async_to_lds_b128 %0, %1, off"
                         :: "v"(ldst), "v"((unsigned long long)gsrc) : "memory");
        }
        if (t0 + 64 < NT) {                        // global_prefetch_b8 next chunk
            __builtin_prefetch(&hidden[(b * NT + t0 + 64) * NC + h * ND + (tid >> 1)], 0, 1);
        }
        // stage 64x128 V chunk, bf16-converted, scattered to fragment layout,
        // packed 2 K-rows per ds_store_b32
#pragma unroll 4
        for (int i = tid; i < 32 * ND; i += 256) { // 16 iters
            int rp = i >> 7;                       // K-row pair 0..31
            int c  = i & 127;
            int r  = rp << 1;
            const float* src = &hidden[(b * NT + t0 + r) * NC + h * ND + c];
            bf2 t; t.x = (__bf16)src[0]; t.y = (__bf16)src[NC];
            int f = r >> 5, kl = r & 31;
            int bl = (((kl >> 3) & 1) << 4) | (c & 15);
            *(bf2*)&frag_b[c >> 4][f][bl][(kl & 7) + ((kl >> 4) << 3)] = t;
        }
        asm volatile("s_wait_asynccnt 0x0" ::: "memory");
        __syncthreads();

        // ---- phase 2: weights tile 16x64 -> A fragments (pairs along K) ----
#pragma unroll
        for (int i = tid; i < 16 * 32; i += 256) { // 2 iters
            int ur  = i >> 5;
            int tc  = (i & 31) << 1;               // even K index 0..62
            float uf = (float)(u0 + ur) + 0.5f;
            float d0 = (uf - align_s[tc])     * sig;
            float d1 = (uf - align_s[tc + 1]) * sig;
            float s0 = -(d0 * d0) + sb;
            float s1 = -(d1 * d1) + sb;
            float w0 = (mask_s[tc]     > 0.f) ? expf(s0 - m_s[ur]) * is_s[ur] : 0.f;
            float w1 = (mask_s[tc + 1] > 0.f) ? expf(s1 - m_s[ur]) * is_s[ur] : 0.f;
            int f = tc >> 5, K = tc & 31;
            int la = ur + (((K >> 3) & 1) << 4);
            int e  = (K & 7) + ((K >> 4) << 3);
            bf2 t; t.x = (__bf16)w0; t.y = (__bf16)w1;
            *(bf2*)&frag_a[f][la][e] = t;
        }
        __syncthreads();

        // ---- phase 3: two independent WMMAs per wave ----
        v16bf a0 = *(const v16bf*)frag_a[0][lane];
        v16bf a1 = *(const v16bf*)frag_a[1][lane];
        v16bf b0 = *(const v16bf*)frag_b[wid][0][lane];
        v16bf b1 = *(const v16bf*)frag_b[wid][1][lane];
        acc0 = __builtin_amdgcn_wmma_f32_16x16x32_bf16(
                   false, a0, false, b0, (short)0, acc0, false, false);
        acc1 = __builtin_amdgcn_wmma_f32_16x16x32_bf16(
                   false, a1, false, b1, (short)0, acc1, false, false);
        __syncthreads();                           // protect LDS vs next stores
    }

    // C/D layout: VGPR i, lanes 0-15 -> M=i, lanes 16-31 -> M=i+8; N=lane%16
    int half = lane >> 4, n = lane & 15;
    int col = h * ND + wid * 16 + n;
#pragma unroll
    for (int i = 0; i < 8; ++i) {
        int u = u0 + i + half * 8;
        if (u < NU) out[(b * NU + u) * NC + col] = acc0[i] + acc1[i];
    }
}

// -------------------------------------------------------------------------
// Host launcher
// -------------------------------------------------------------------------
extern "C" void kernel_launch(void* const* d_in, const int* in_sizes, int n_in,
                              void* d_out, int out_size, void* d_ws, size_t ws_size,
                              hipStream_t stream)
{
    const float* hidden     = (const float*)d_in[0];
    const float* mask       = (const float*)d_in[1];
    const int*   tgt_len    = (const int*)  d_in[2];
    // d_in[3] = max_token (compile-time NU)
    const float* conv_w     = (const float*)d_in[4];
    const float* conv_b     = (const float*)d_in[5];
    const float* lin_w      = (const float*)d_in[6];
    const float* lin_b      = (const float*)d_in[7];
    const float* sigma      = (const float*)d_in[8];
    const float* sigma_bias = (const float*)d_in[9];
    (void)in_sizes; (void)n_in; (void)out_size; (void)ws_size;

    float* wsf   = (float*)d_ws;
    float* raw   = wsf;                         // [NB*NT]
    float* align = wsf + NB * NT;               // [NB*NT]
    float* mrow  = wsf + 2 * NB * NT;           // [NB*NH*NU]
    float* isrow = mrow + NB * NH * NU;         // [NB*NH*NU]

    float* out        = (float*)d_out;
    float* out_token  = out + NB * NU * NC;     // acoustic | token_num | alphas
    float* out_alphas = out_token + NB;

    alpha_kernel<<<(NB * NT * 32) / 256, 256, 0, stream>>>(
        hidden, mask, conv_w, conv_b, lin_w, lin_b, raw);

    scan_kernel<<<NB, 256, 0, stream>>>(
        raw, tgt_len, align, out_token, out_alphas);

    rowstat_kernel<<<(NB * NH * NU * 32) / 256, 256, 0, stream>>>(
        align, mask, sigma, sigma_bias, mrow, isrow);

    attn_wmma_kernel<<<NB * NH * 16, 256, 0, stream>>>(
        hidden, mask, align, mrow, isrow, sigma, sigma_bias, out);
}
